// DeepSeekV3Router_26534307954918
// MI455X (gfx1250) — matrix operand order repaired and checked by reading, not compiled, not run on hardware
//
#include <hip/hip_runtime.h>
#include <math.h>

#define T_TOKENS    16384
#define D_MODEL     2048
#define NUM_EXPERTS 256
#define TOP_K       8
#define ALPHA_C     0.001f

typedef unsigned short us16;
typedef unsigned int   u32;
typedef __attribute__((ext_vector_type(16))) __bf16 v16bf;
typedef __attribute__((ext_vector_type(8)))  float  v8f;
typedef __attribute__((ext_vector_type(4)))  int    v4i;

struct U32x8 { uint4 a, b; };   // 32 bytes == sizeof(v16bf)

// ---------------- f32 <-> split-bf16 helpers ----------------
__device__ __forceinline__ u32 f32_to_bf16_u(float f) {
    u32 u = __float_as_uint(f);
    u += 0x7FFFu + ((u >> 16) & 1u);   // round-to-nearest-even
    return u >> 16;
}
__device__ __forceinline__ float bf16u_to_f32(u32 h) {
    return __uint_as_float(h << 16);
}
__device__ __forceinline__ void split2(float x0, float x1, u32& hi, u32& lo) {
    u32 h0 = f32_to_bf16_u(x0);
    u32 h1 = f32_to_bf16_u(x1);
    u32 l0 = f32_to_bf16_u(x0 - bf16u_to_f32(h0));
    u32 l1 = f32_to_bf16_u(x1 - bf16u_to_f32(h1));
    hi = h0 | (h1 << 16);
    lo = l0 | (l1 << 16);
}

// ---------------- async global->LDS copy (gfx1250), with fallback ----------------
#if defined(__has_builtin)
#if __has_builtin(__builtin_amdgcn_global_load_async_to_lds_b128)
#define HAS_ASYNC_COPY 1
#endif
#endif
#ifndef HAS_ASYNC_COPY
#define HAS_ASYNC_COPY 0
#endif

#if HAS_ASYNC_COPY
#define N_STAGES 2
__device__ __forceinline__ void async_cp16(void* lds, const void* g) {
    const __attribute__((address_space(1))) void* g1 =
        (const __attribute__((address_space(1))) void*)g;
    __attribute__((address_space(3))) void* l3 =
        (__attribute__((address_space(3))) void*)lds;
    __builtin_amdgcn_global_load_async_to_lds_b128(
        (__attribute__((address_space(1))) v4i*)g1,
        (__attribute__((address_space(3))) v4i*)l3, 0, 0);
}
__device__ __forceinline__ void wait_async0() {
#if __has_builtin(__builtin_amdgcn_s_wait_asynccnt)
    __builtin_amdgcn_s_wait_asynccnt(0);
#else
    asm volatile("s_wait_asynccnt 0x0" ::: "memory");
#endif
}
#else
#define N_STAGES 1
#endif

// ---------------- pre-split: f32 -> bf16 hi/lo planes ----------------
__global__ __launch_bounds__(256)
void split_planes(const float* __restrict__ x, us16* __restrict__ hi,
                  us16* __restrict__ lo, int n4)
{
    int i = blockIdx.x * 256 + threadIdx.x;
    if (i >= n4) return;
    float4 f = ((const float4*)x)[i];
    u32 h0, l0, h1, l1;
    split2(f.x, f.y, h0, l0);
    split2(f.z, f.w, h1, l1);
    uint2 H; H.x = h0; H.y = h1;
    uint2 L; L.x = l0; L.y = l1;
    ((uint2*)hi)[i] = H;
    ((uint2*)lo)[i] = L;
}

// ---------------- NT GEMM: C = scale * A[M,K] * B[N,K]^T  (bf16x3, f32 acc) ----
// A,B given as pre-split bf16 hi/lo planes. Output: f32 (split_out=0) or
// bf16 hi/lo planes (split_out=1, feeds the next GEMM's A operand).
#define BM 128
#define BN 128
#define BK 64
// LDS: [stage][plane(Ahi,Alo,Bhi,Blo)][row*8 + swizzled-chunk], chunk = 16B = 8 bf16
__global__ __launch_bounds__(256)
void gemm_nt_bf16x3(const us16* __restrict__ Ahi, const us16* __restrict__ Alo,
                    const us16* __restrict__ Bhi, const us16* __restrict__ Blo,
                    float* __restrict__ Cf, us16* __restrict__ Chi,
                    us16* __restrict__ Clo, int M, int N, int K,
                    float scale, int split_out)
{
    __shared__ uint4 sm[N_STAGES][4][BM * 8];

    const int tid  = threadIdx.x;
    const int lane = tid & 31;
    const int wav  = tid >> 5;
    const int wm   = wav >> 2;      // 0..1 (64-row strip)
    const int wn   = wav & 3;       // 0..3 (32-col strip)
    const int hf   = lane >> 4;     // 0/1
    const int lr   = lane & 15;

    const int m0 = blockIdx.y * BM;
    const int n0 = blockIdx.x * BN;

    // fragment chunk indices into sm[..][plane][..] (XOR bank swizzle on 8 chunks)
    int iA0[2][4], iA1[2][4], iB0[2][2], iB1[2][2];
    #pragma unroll
    for (int i = 0; i < 4; ++i) {
        int row = wm * 64 + i * 16 + lr;
        int sw  = (row >> 1) & 7;
        #pragma unroll
        for (int ks = 0; ks < 2; ++ks) {
            iA0[ks][i] = row * 8 + ((ks * 4 + hf) ^ sw);       // K +0..7  / +8..15
            iA1[ks][i] = row * 8 + ((ks * 4 + 2 + hf) ^ sw);   // K +16..23 / +24..31
        }
    }
    #pragma unroll
    for (int j = 0; j < 2; ++j) {
        int row = wn * 32 + j * 16 + lr;       // B^T column == B row
        int sw  = (row >> 1) & 7;
        #pragma unroll
        for (int ks = 0; ks < 2; ++ks) {
            iB0[ks][j] = row * 8 + ((ks * 4 + hf) ^ sw);       // K +0..7  / +8..15
            iB1[ks][j] = row * 8 + ((ks * 4 + 2 + hf) ^ sw);   // K +16..23 / +24..31
        }
    }

    v8f acc[4][2];
    #pragma unroll
    for (int i = 0; i < 4; ++i)
        #pragma unroll
        for (int j = 0; j < 2; ++j)
            #pragma unroll
            for (int q = 0; q < 8; ++q) acc[i][j][q] = 0.0f;

#define DO_STAGE(SMP) do {                                                      \
    _Pragma("unroll") for (int ks = 0; ks < 2; ++ks) {                          \
        v16bf afh[4], afl[4], bfh[2], bfl[2];                                   \
        _Pragma("unroll") for (int i = 0; i < 4; ++i) { U32x8 u;                \
            u.a = (SMP)[0][iA0[ks][i]]; u.b = (SMP)[0][iA1[ks][i]];             \
            afh[i] = __builtin_bit_cast(v16bf, u);                              \
            u.a = (SMP)[1][iA0[ks][i]]; u.b = (SMP)[1][iA1[ks][i]];             \
            afl[i] = __builtin_bit_cast(v16bf, u); }                            \
        _Pragma("unroll") for (int j = 0; j < 2; ++j) { U32x8 u;                \
            u.a = (SMP)[2][iB0[ks][j]]; u.b = (SMP)[2][iB1[ks][j]];             \
            bfh[j] = __builtin_bit_cast(v16bf, u);                              \
            u.a = (SMP)[3][iB0[ks][j]]; u.b = (SMP)[3][iB1[ks][j]];             \
            bfl[j] = __builtin_bit_cast(v16bf, u); }                            \
        _Pragma("unroll") for (int i = 0; i < 4; ++i)                           \
            _Pragma("unroll") for (int j = 0; j < 2; ++j) {                     \
                acc[i][j] = __builtin_amdgcn_wmma_f32_16x16x32_bf16(            \
                    false, afh[i], false, bfh[j], (short)0, acc[i][j], false, false); \
                acc[i][j] = __builtin_amdgcn_wmma_f32_16x16x32_bf16(            \
                    false, afh[i], false, bfl[j], (short)0, acc[i][j], false, false); \
                acc[i][j] = __builtin_amdgcn_wmma_f32_16x16x32_bf16(            \
                    false, afl[i], false, bfh[j], (short)0, acc[i][j], false, false); \
            }                                                                   \
    }                                                                           \
} while (0)

#if HAS_ASYNC_COPY
#define ISSUE_STAGE(DSTP, KK) do {                                              \
    _Pragma("unroll") for (int v = 0; v < 4; ++v) {                             \
        int c = tid + 256 * v;                                                  \
        int row = c >> 3, ch = c & 7;                                           \
        int chs = ch ^ ((row >> 1) & 7);                                        \
        size_t ga = (size_t)(m0 + row) * K + (KK) + ch * 8;                     \
        size_t gb = (size_t)(n0 + row) * K + (KK) + ch * 8;                     \
        async_cp16((void*)&(DSTP)[0][row * 8 + chs], Ahi + ga);                 \
        async_cp16((void*)&(DSTP)[1][row * 8 + chs], Alo + ga);                 \
        async_cp16((void*)&(DSTP)[2][row * 8 + chs], Bhi + gb);                 \
        async_cp16((void*)&(DSTP)[3][row * 8 + chs], Blo + gb);                 \
    }                                                                           \
} while (0)

    // software pipeline: one barrier + one asynccnt wait per 48 WMMAs
    ISSUE_STAGE(sm[0], 0);
    int cur = 0;
    for (int kk = 0; kk < K - BK; kk += BK) {
        wait_async0();        // my stage-`cur` async writes have landed in LDS
        __syncthreads();      // everyone's landed; everyone done with buf cur^1
        ISSUE_STAGE(sm[cur ^ 1], kk + BK);   // unconditional (last step peeled)
        DO_STAGE(sm[cur]);
        cur ^= 1;
    }
    wait_async0();
    __syncthreads();
    DO_STAGE(sm[cur]);
#else
    for (int kk = 0; kk < K; kk += BK) {
        __syncthreads();
        #pragma unroll
        for (int v = 0; v < 4; ++v) {
            int c = tid + 256 * v;
            int row = c >> 3, ch = c & 7;
            int chs = ch ^ ((row >> 1) & 7);
            size_t ga = (size_t)(m0 + row) * K + kk + ch * 8;
            size_t gb = (size_t)(n0 + row) * K + kk + ch * 8;
            sm[0][0][row * 8 + chs] = *(const uint4*)(Ahi + ga);
            sm[0][1][row * 8 + chs] = *(const uint4*)(Alo + ga);
            sm[0][2][row * 8 + chs] = *(const uint4*)(Bhi + gb);
            sm[0][3][row * 8 + chs] = *(const uint4*)(Blo + gb);
        }
        __syncthreads();
        DO_STAGE(sm[0]);
    }
#endif

    // epilogue: VGPR r of C/D holds M = r + 8*hf, N = lr (per 16x16 tile)
    #pragma unroll
    for (int i = 0; i < 4; ++i)
        #pragma unroll
        for (int j = 0; j < 2; ++j) {
            int col  = n0 + wn * 32 + j * 16 + lr;
            int rowb = m0 + wm * 64 + i * 16 + hf * 8;
            #pragma unroll
            for (int r = 0; r < 8; ++r) {
                float val = acc[i][j][r] * scale;
                size_t off = (size_t)(rowb + r) * N + col;
                if (split_out) {
                    u32 h = f32_to_bf16_u(val);
                    u32 l = f32_to_bf16_u(val - bf16u_to_f32(h));
                    Chi[off] = (us16)h;
                    Clo[off] = (us16)l;
                } else {
                    Cf[off] = val;
                }
            }
        }
#undef DO_STAGE
#undef ISSUE_STAGE
}

// ---------------- router: sigmoid, biased top-8, weights, loss partials ------
__global__ __launch_bounds__(256)
void router_topk(const float* __restrict__ aff, const float* __restrict__ bias,
                 float* __restrict__ idx_out, float* __restrict__ w_out,
                 float* __restrict__ counts, float* __restrict__ psum)
{
    __shared__ float lds_psum[NUM_EXPERTS];
    const int tid   = threadIdx.x;
    const int lane  = tid & 31;
    const int wav   = tid >> 5;
    const int token = blockIdx.x * 8 + wav;

    if (tid < NUM_EXPERTS) lds_psum[tid] = 0.0f;
    __syncthreads();

    float gate[8], sel[8];
    #pragma unroll
    for (int j = 0; j < 8; ++j) {
        int e = lane + 32 * j;
        float a = aff[(size_t)token * NUM_EXPERTS + e];
        float g = 1.0f / (1.0f + expf(-a));
        gate[j] = g;
        sel[j]  = g + bias[e];
    }

    float my_g = 0.0f; int my_i = 0;
    float gsum = 0.0f;
    for (int k = 0; k < TOP_K; ++k) {
        float bv = sel[0]; int bi = lane;
        #pragma unroll
        for (int j = 1; j < 8; ++j) {
            int e = lane + 32 * j;
            if (sel[j] > bv) { bv = sel[j]; bi = e; }
        }
        #pragma unroll
        for (int off = 16; off >= 1; off >>= 1) {
            float ov = __shfl_xor(bv, off, 32);
            int   oi = __shfl_xor(bi, off, 32);
            if (ov > bv || (ov == bv && oi < bi)) { bv = ov; bi = oi; }
        }
        float g = bv - bias[bi];          // unbiased gate of winner
        int srcLane = bi & 31;
        int slot    = bi >> 5;
        #pragma unroll
        for (int j = 0; j < 8; ++j)
            if (lane == srcLane && slot == j) sel[j] = -INFINITY;
        if (lane == k) { my_g = g; my_i = bi; }
        gsum += g;
    }

    if (lane < TOP_K) {
        idx_out[(size_t)token * TOP_K + lane] = (float)my_i;
        w_out  [(size_t)token * TOP_K + lane] = my_g / (gsum + 1e-10f);
        atomicAdd(&counts[my_i], 1.0f);
    }
    #pragma unroll
    for (int j = 0; j < 8; ++j)
        atomicAdd(&lds_psum[lane + 32 * j], gate[j]);
    __syncthreads();
    if (tid < NUM_EXPERTS) atomicAdd(&psum[tid], lds_psum[tid]);
}

__global__ __launch_bounds__(256)
void router_loss(const float* __restrict__ counts, const float* __restrict__ psum,
                 float* __restrict__ loss_out)
{
    __shared__ float red[NUM_EXPERTS];
    int t = threadIdx.x;
    float f_e = counts[t] / (float)(T_TOKENS * TOP_K);
    float P_e = psum[t] / (float)T_TOKENS;
    red[t] = f_e * P_e;
    __syncthreads();
    for (int s = NUM_EXPERTS / 2; s > 0; s >>= 1) {
        if (t < s) red[t] += red[t + s];
        __syncthreads();
    }
    if (t == 0)
        *loss_out = ALPHA_C * ((float)NUM_EXPERTS / (float)TOP_K) * red[0];
}

// ---------------- host ----------------
extern "C" void kernel_launch(void* const* d_in, const int* in_sizes, int n_in,
                              void* d_out, int out_size, void* d_ws, size_t ws_size,
                              hipStream_t stream)
{
    const float* hidden    = (const float*)d_in[0];  // [T, D]
    const float* W_proj    = (const float*)d_in[1];  // [D, D]
    const float* centroids = (const float*)d_in[2];  // [E, D]
    const float* bias      = (const float*)d_in[3];  // [E]

    float* out      = (float*)d_out;
    float* idx_out  = out;                                        // [T, 8]
    float* w_out    = out + (size_t)T_TOKENS * TOP_K;             // [T, 8]
    float* aff_out  = out + (size_t)T_TOKENS * TOP_K * 2;         // [T, 256]
    float* loss_out = aff_out + (size_t)T_TOKENS * NUM_EXPERTS;   // [1]

    const size_t TD = (size_t)T_TOKENS * D_MODEL;
    const size_t DD = (size_t)D_MODEL * D_MODEL;
    const size_t ED = (size_t)NUM_EXPERTS * D_MODEL;

    float* counts = (float*)d_ws;                  // [256]
    float* psum   = counts + NUM_EXPERTS;          // [256]
    us16*  planes = (us16*)((float*)d_ws + 1024);  // 4KB offset
    us16* hHi = planes;     us16* hLo = hHi + TD;
    us16* wHi = hLo + TD;   us16* wLo = wHi + DD;
    us16* cHi = wLo + DD;   us16* cLo = cHi + ED;
    us16* tHi = cLo + ED;   us16* tLo = tHi + TD;  // token_features planes

    (void)hipMemsetAsync(d_ws, 0, 2 * NUM_EXPERTS * sizeof(float), stream);

    dim3 blk(256);
    // one-time f32 -> bf16 hi/lo splits (memory bound, done once per operand)
    split_planes<<<dim3((unsigned)(TD / 4 / 256)), blk, 0, stream>>>(hidden,    hHi, hLo, (int)(TD / 4));
    split_planes<<<dim3((unsigned)(DD / 4 / 256)), blk, 0, stream>>>(W_proj,    wHi, wLo, (int)(DD / 4));
    split_planes<<<dim3((unsigned)(ED / 4 / 256)), blk, 0, stream>>>(centroids, cHi, cLo, (int)(ED / 4));

    // token_features = hidden @ W_proj^T  (written directly as hi/lo planes)
    gemm_nt_bf16x3<<<dim3(D_MODEL / BN, T_TOKENS / BM), blk, 0, stream>>>(
        hHi, hLo, wHi, wLo, nullptr, tHi, tLo,
        T_TOKENS, D_MODEL, D_MODEL, 1.0f, 1);
    // affinities = token_features @ centroids^T / TEMPERATURE (T=1.0)
    gemm_nt_bf16x3<<<dim3(NUM_EXPERTS / BN, T_TOKENS / BM), blk, 0, stream>>>(
        tHi, tLo, cHi, cLo, aff_out, nullptr, nullptr,
        T_TOKENS, NUM_EXPERTS, D_MODEL, 1.0f, 0);

    router_topk<<<dim3(T_TOKENS / 8), blk, 0, stream>>>(
        aff_out, bias, idx_out, w_out, counts, psum);
    router_loss<<<dim3(1), blk, 0, stream>>>(counts, psum, loss_out);
}
